// Attention_20109036880333
// MI455X (gfx1250) — matrix operand order repaired and checked by reading, not compiled, not run on hardware
//
#include <hip/hip_runtime.h>

// ---------------------------------------------------------------------------
// Causal flash attention for MI455X (gfx1250, wave32, WMMA 16x16x32 f16).
//
// B=8, S=2048, D=128, fp32 in/out.  Matrix-core bound (8.6 GFLOP causal vs
// ~34MB of HBM traffic), so everything runs on v_wmma_f32_16x16x32_f16 with
// f32 accumulation.
//   * pre-pass: Q,K -> f16 row-major; V -> group-blocked packed-pair f16
//     Vp[b][g][d][j] = {V[g*16+2j][d], V[g*16+2j+1][d]}  (g = 16-key group)
//     so a V B-operand fragment is two B128 loads (pairs contiguous).
//   * main kernel: 4 waves/block, 16 queries/wave, 32 keys/iteration.
//     Scores computed transposed (S^T = K * Q^T) so softmax is a per-lane
//     reduction; S^T's D-layout coincides with the f16 A-operand layout of P,
//     so P*V needs only v_cvt_pk packs (no LDS, no shuffles).
//   * causal balance: heaviest (diagonal-most) query tiles launch first so
//     trailing light tiles fill the WGP tail.
// ---------------------------------------------------------------------------

typedef __attribute__((ext_vector_type(16))) _Float16 v16h;
typedef __attribute__((ext_vector_type(8)))  float    v8f;
typedef __attribute__((ext_vector_type(2)))  __fp16   v2fp16;
typedef __attribute__((ext_vector_type(4)))  unsigned v4u;
typedef __attribute__((ext_vector_type(4)))  float    v4f;
typedef __attribute__((ext_vector_type(2)))  float    v2f;

#define B_N   8
#define S_LEN 2048
#define D_DIM 128

union Frag {
  v16h h;
  v4u  q[2];
  unsigned u[8];
};

static __device__ __forceinline__ unsigned pkrtz(float a, float b) {
  v2fp16 t = __builtin_amdgcn_cvt_pkrtz(a, b);   // v_cvt_pk_rtz_f16_f32
  return __builtin_bit_cast(unsigned, t);
}

static __device__ __forceinline__ float fexp2(float x) {
#if __has_builtin(__builtin_amdgcn_exp2f)
  return __builtin_amdgcn_exp2f(x);              // v_exp_f32
#else
  return exp2f(x);
#endif
}

// ---------------- pre-pass: fp32 -> fp16 row-major (Q and K) ---------------
__global__ void cvt_f16_kernel(const float* __restrict__ in,
                               _Float16* __restrict__ out) {
  size_t i = ((size_t)blockIdx.x * blockDim.x + threadIdx.x) * 2;
  v2f v = *(const v2f*)(in + i);
  *(unsigned*)(out + i) = pkrtz(v.x, v.y);
}

// ---- pre-pass: V -> Vp[b][g][d][j] = pack(V[g*16+2j][d], V[g*16+2j+1][d]) --
// One block per (b,g): stage the 16x128 f32 tile in LDS (coalesced read),
// emit the 128x8 packed-u32 tile (coalesced 16B writes).
__global__ __launch_bounds__(256)
void pack_v_kernel(const float* __restrict__ V, unsigned* __restrict__ Vp) {
  __shared__ float lds[16 * D_DIM];              // 8 KB
  const int t = threadIdx.x;
  const size_t bg = blockIdx.x;                  // b*(S/16) + g

  const v4f* s4 = (const v4f*)(V + bg * (16 * D_DIM));
  v4f* l4 = (v4f*)lds;
  l4[t * 2]     = s4[t * 2];
  l4[t * 2 + 1] = s4[t * 2 + 1];
  __syncthreads();

  v4u r;
#pragma unroll
  for (int q = 0; q < 4; ++q) {
    int o = t * 4 + q;                           // 0..1023
    int j = o & 7;                               // key-pair in group
    int d = o >> 3;                              // 0..127
    r[q] = pkrtz(lds[(2 * j) * D_DIM + d], lds[(2 * j + 1) * D_DIM + d]);
  }
  ((v4u*)(Vp + bg * (D_DIM * 8)))[t] = r;
}

// ------------------------------ main kernel --------------------------------
__global__ __launch_bounds__(128, 1)
void flash_attn_causal(const _Float16* __restrict__ Qh,
                       const _Float16* __restrict__ Kh,
                       const unsigned* __restrict__ Vp,
                       float* __restrict__ Out) {
  const int lane = threadIdx.x & 31;
  const int wave = threadIdx.x >> 5;
  const int n16  = lane & 15;       // q column (scores) / d column (output)
  const int half = lane >> 4;       // lane half (0/1)
  const int b    = blockIdx.y;
  // Heavy diagonal tiles first: reverse the query-tile order.
  const int qtile = (int)gridDim.x - 1 - (int)blockIdx.x;
  const int qbw   = qtile * 64 + wave * 16;       // this wave's first query

  const float c2 = 0.08838834764831845f * 1.4426950408889634f; // log2(e)/sqrt(D)

  // Q^T B-operand fragments (one per 32-wide d chunk), loaded once.
  // B layout: reg v holds d = 2v,2v+1 (+8 for lanes 16-31, +16 for regs 4-7), lane%16 = q.
  const _Float16* qrow = Qh + ((size_t)b * S_LEN + qbw + n16) * D_DIM;
  Frag qf[4];
#pragma unroll
  for (int c = 0; c < 4; ++c) {
    const _Float16* p = qrow + c * 32 + half * 8;
    qf[c].q[0] = *(const v4u*)(p);
    qf[c].q[1] = *(const v4u*)(p + 16);
  }

  v8f acc[8];                       // O tile, (16q x 128d) in 8 d-chunks
#pragma unroll
  for (int c = 0; c < 8; ++c) acc[c] = (v8f)0.0f;

  float m = -3.0e38f;               // running row max (per query column)
  float l = 0.0f;                   // running row sum (this lane-half's keys)
  const int q_glob = qbw + n16;

  for (int kb = 0; kb <= qbw + 15; kb += 32) {
    const bool do1 = (kb + 16) <= (qbw + 15);   // upper 16-key half needed?

    const _Float16* krow0 = Kh + ((size_t)b * S_LEN + kb + n16) * D_DIM;
    __builtin_prefetch((const void*)(krow0 + 32 * D_DIM), 0, 1); // next K tile

    // ---- scores S^T = K * Q^T : two 16(key)x16(query) f32 tiles ----
    v8f s0 = (v8f)0.0f, s1 = (v8f)0.0f;
#pragma unroll
    for (int c = 0; c < 4; ++c) {
      Frag kf;
      const _Float16* p = krow0 + c * 32 + half * 8;
      kf.q[0] = *(const v4u*)(p);          // A: K rows, d pairs contiguous
      kf.q[1] = *(const v4u*)(p + 16);
      s0 = __builtin_amdgcn_wmma_f32_16x16x32_f16(false, kf.h, false, qf[c].h,
                                                  (short)0, s0, false, false);
    }
    if (do1) {
      const _Float16* krow1 = krow0 + 16 * D_DIM;
#pragma unroll
      for (int c = 0; c < 4; ++c) {
        Frag kf;
        const _Float16* p = krow1 + c * 32 + half * 8;
        kf.q[0] = *(const v4u*)(p);
        kf.q[1] = *(const v4u*)(p + 16);
        s1 = __builtin_amdgcn_wmma_f32_16x16x32_f16(false, kf.h, false, qf[c].h,
                                                    (short)0, s1, false, false);
      }
    } else {
#pragma unroll
      for (int r = 0; r < 8; ++r) s1[r] = -3.0e38f;
    }

    // ---- causal mask (only the diagonal tile takes this branch) ----
    if (kb + 31 > qbw) {
#pragma unroll
      for (int r = 0; r < 8; ++r) {
        int k0 = kb + r + half * 8;           // key row of reg r (tile 0)
        if (k0 > q_glob)      s0[r] = -3.0e38f;
        if (k0 + 16 > q_glob) s1[r] = -3.0e38f;
      }
    }

    // ---- online softmax: rows live along VGPRs -> per-lane reduce ----
    float tmax = fmaxf(s0[0], s1[0]);
#pragma unroll
    for (int r = 1; r < 8; ++r) tmax = fmaxf(tmax, fmaxf(s0[r], s1[r]));
    tmax = fmaxf(tmax, __shfl_xor(tmax, 16, 32));   // combine lane halves
    const float mold = m;
    const float mnew = fmaxf(m, tmax);
    const float alpha = fexp2((mold - mnew) * c2);
    m = mnew;

    // p = exp2((s - m) * log2e/sqrt(D)); pack straight into the A-operand of
    // P*V (S^T D-layout == P A-layout: lane%16 = q, reg pairs = key pairs).
    float tsum = 0.0f;
    Frag pf;
#pragma unroll
    for (int r = 0; r < 4; ++r) {
      float p0a = fexp2((s0[2 * r]     - mnew) * c2);
      float p0b = fexp2((s0[2 * r + 1] - mnew) * c2);
      float p1a = fexp2((s1[2 * r]     - mnew) * c2);
      float p1b = fexp2((s1[2 * r + 1] - mnew) * c2);
      tsum += (p0a + p0b) + (p1a + p1b);
      pf.u[r]     = pkrtz(p0a, p0b);
      pf.u[4 + r] = pkrtz(p1a, p1b);
    }
    l = l * alpha + tsum;

    // Rescale O only when some lane's max actually moved (wave-uniform vote;
    // EXEC stays all-ones, required around WMMA).  alpha per O-layout row
    // (q = r + half*8) is broadcast from lane q.
    if (!__all(mnew == mold)) {
      float ar[8];
#pragma unroll
      for (int r = 0; r < 8; ++r) ar[r] = __shfl(alpha, r + half * 8, 32);
#pragma unroll
      for (int c = 0; c < 8; ++c)
#pragma unroll
        for (int r = 0; r < 8; ++r) acc[c][r] *= ar[r];
    }

    // ---- O += P * V ----
    // Vp[b][g][d][j]: this lane-half needs j = 4*half..4*half+3 of groups
    // g = kb/16 and kb/16+1 -> two B128 loads per d-chunk.
    const unsigned* vb = Vp +
        ((size_t)b * (S_LEN / 16) + (kb >> 4)) * (D_DIM * 8);
    __builtin_prefetch((const void*)(vb + 2 * D_DIM * 8), 0, 1); // next V tile
#pragma unroll
    for (int c = 0; c < 8; ++c) {
      Frag vf;                       // B: reg v = key pair (+8 for lanes 16-31, +16 for regs 4-7), lane%16 = d
      const v4u* vp4 = (const v4u*)(vb + (size_t)(c * 16 + n16) * 8 + half * 4);
      vf.q[0] = vp4[0];
      vf.q[1] = vp4[2 * D_DIM];
      acc[c] = __builtin_amdgcn_wmma_f32_16x16x32_f16(false, pf.h, false, vf.h,
                                                      (short)0, acc[c],
                                                      false, false);
    }
  }

  // ---- finalize: 1/l per query row, coalesced stores ----
  l += __shfl_xor(l, 16, 32);
  float linv = 1.0f / l;
  float lr[8];
#pragma unroll
  for (int r = 0; r < 8; ++r) lr[r] = __shfl(linv, r + half * 8, 32);

  float* orow = Out + ((size_t)b * S_LEN + qbw + half * 8) * D_DIM + n16;
#pragma unroll
  for (int c = 0; c < 8; ++c)
#pragma unroll
    for (int r = 0; r < 8; ++r)
      orow[(size_t)r * D_DIM + c * 16] = acc[c][r] * lr[r];
}

// ---------------------------------------------------------------------------
extern "C" void kernel_launch(void* const* d_in, const int* in_sizes, int n_in,
                              void* d_out, int out_size, void* d_ws,
                              size_t ws_size, hipStream_t stream) {
  (void)in_sizes; (void)n_in; (void)out_size; (void)ws_size;
  const float* Q = (const float*)d_in[0];
  const float* K = (const float*)d_in[1];
  const float* V = (const float*)d_in[2];
  float* Out = (float*)d_out;

  const size_t N = (size_t)B_N * S_LEN * D_DIM;   // 2,097,152 elements
  _Float16* Qh = (_Float16*)d_ws;                 // 4 MB
  _Float16* Kh = Qh + N;                          // 4 MB
  unsigned* Vp = (unsigned*)(Kh + N);             // 4 MB (needs ws >= 12 MB)

  const int cb = (int)(N / 2 / 256);              // 4096 blocks
  cvt_f16_kernel<<<cb, 256, 0, stream>>>(Q, Qh);
  cvt_f16_kernel<<<cb, 256, 0, stream>>>(K, Kh);
  pack_v_kernel<<<B_N * (S_LEN / 16), 256, 0, stream>>>(V, Vp);

  dim3 grid(S_LEN / 64, B_N);                     // 32 query-tiles x 8 batches
  flash_attn_causal<<<grid, 128, 0, stream>>>(Qh, Kh, Vp, Out);
}
